// InvariantPointAttention_43052752175460
// MI455X (gfx1250) — matrix operand order repaired and checked by reading, not compiled, not run on hardware
//
#include <hip/hip_runtime.h>
#include <hip/hip_bf16.h>
#include <math.h>

// ---------------------------------------------------------------------------
// InvariantPointAttention for MI455X (gfx1250, wave32, WMMA bf16 16x16x32)
//
// Shapes: N=512, C_S=384, C_Z=128, C_H=16, H=12, PQ=4, PV=8, CAT=2112
// Memory-bound on z_ij (134MB). z is streamed twice: once by the b_ij WMMA
// pass, once by the fused flash attention (second pass largely L2-resident,
// 192MB L2 > 134MB z). Inside the attention kernel the per-tile z slab is
// staged into LDS by the Tensor Data Mover (tensor_load_to_lds, 3-D D#,
// s_wait_tensorcnt) so the o_hat B operands come from conflict-free LDS
// instead of strided VMEM, and the DMA overlaps the logits/softmax phase.
// ---------------------------------------------------------------------------

typedef __attribute__((ext_vector_type(16))) __bf16 v16bf;
typedef __attribute__((ext_vector_type(8)))  __bf16 v8bf;
typedef __attribute__((ext_vector_type(8)))  float  v8f;
typedef __attribute__((ext_vector_type(4)))  unsigned int v4u;
typedef __attribute__((ext_vector_type(8)))  int v8i;
typedef __attribute__((ext_vector_type(4)))  int v4i;

union U16 { v16bf v; v8bf h[2]; };

#define N_    512
#define C_S_  384
#define C_Z_  128
#define C_H_  16
#define H_    12
#define PQ_   4
#define PV_   8
#define CAT_W 2112
#define W_Lc  0.57735026918962576f   /* sqrt(1/3) */
#define W_Cc  0.23570226039551584f   /* sqrt(2/36) */

__device__ __forceinline__ v8f v8f_zero() {
  v8f r; for (int i = 0; i < 8; ++i) r[i] = 0.f; return r;
}

__device__ __forceinline__ v8f wmma_bf16(const U16& a, const U16& b, v8f c) {
  return __builtin_amdgcn_wmma_f32_16x16x32_bf16(false, a.v, false, b.v,
                                                 (short)0, c, false, false);
}

// TDM: load 3-D tile z[i:16][j:32][d:128] f32 -> LDS (contiguous), D# per
// CDNA5 ISA ch.8. tensor dims: dim0=128 d, dim1=j lines (stride 128 elems),
// dim2=i lines (stride 65536 elems). This toolchain's builtin is the 6-arg
// form: (g0 v4u, g1 v8i, g2 v4i, g3 v4i, extra v8i, cpol).
__device__ __forceinline__ void tdm_load_z_slab(const float* gptr,
                                                unsigned lds_off) {
  const unsigned long long ga = (unsigned long long)(uintptr_t)gptr;
  v4u g0;
  g0[0] = 1u;                                   // count=1, user descriptor
  g0[1] = lds_off;                              // lds_addr (bytes)
  g0[2] = (unsigned)(ga & 0xffffffffu);         // global_addr[31:0]
  g0[3] = (unsigned)((ga >> 32) & 0x01ffffffu)  // global_addr[56:32]
        | (2u << 30);                           // type = 2 ("image")
  v8i g1;
  g1[0] = (int)(2u << 16);                      // data_size = 4 bytes
  g1[1] = (int)(128u << 16);                    // tensor_dim0 = 128 (lo16)
  g1[2] = 0;                                    // dim0 hi | dim1 lo16 = 0
  g1[3] = (int)((262144u >> 16) | (128u << 16));// dim1 hi | tile_dim0 = 128
  g1[4] = (int)(32u | (16u << 16));             // tile_dim1=32 j, tile_dim2=16 i
  g1[5] = 128;                                  // tensor_dim0_stride = 128
  g1[6] = 0;                                    // d0str hi | d1str lo16
  g1[7] = 1;                                    // d1str[47:16]=1 -> 65536
  v4i g2;
  g2[0] = 512;                                  // tensor_dim2 = 512 i lines
  g2[1] = 0; g2[2] = 0; g2[3] = 0;              // dim3 unused, tile_dim3 = 0
  v4i g3;
  g3[0] = 0; g3[1] = 0; g3[2] = 0; g3[3] = 0;   // dim4 unused
  v8i g4;
  for (int k = 0; k < 8; ++k) g4[k] = 0;        // trailing group (unused dims)
  __builtin_amdgcn_tensor_load_to_lds(g0, g1, g2, g3, g4, 0);
}

// ---------------------------------------------------------------------------
// small prep kernels
// ---------------------------------------------------------------------------
__global__ void cvt_bf16_kernel(__bf16* __restrict__ dst,
                                const float* __restrict__ src, int n) {
  int i = blockIdx.x * blockDim.x + threadIdx.x;
  if (i < n) dst[i] = (__bf16)src[i];
}

// dst[c][r] = (bf16) src[r][c]    (rows x cols input)
__global__ void transpose_cvt_kernel(__bf16* __restrict__ dst,
                                     const float* __restrict__ src,
                                     int rows, int cols) {
  int i = blockIdx.x * blockDim.x + threadIdx.x;
  if (i >= rows * cols) return;
  int r = i / cols, c = i % cols;
  dst[(size_t)c * rows + r] = (__bf16)src[i];
}

// WbT[16][128] with W_L folded; rows 12..15 zero (N pad for WMMA)
__global__ void prep_wb_kernel(__bf16* __restrict__ WbT,
                               const float* __restrict__ Wb) {
  int i = blockIdx.x * blockDim.x + threadIdx.x;
  if (i >= 16 * C_Z_) return;
  int h = i / C_Z_, d = i % C_Z_;
  WbT[i] = (h < H_) ? (__bf16)(W_Lc * Wb[d * H_ + h]) : (__bf16)0.f;
}

// ---------------------------------------------------------------------------
// generic bf16 WMMA GEMM:  C[M x N] = A[M x K] * Bt[N x K]^T (+bias)
// one wave per 16x16 C tile; K multiple of 32.
// ---------------------------------------------------------------------------
__global__ void gemm_bf16_kernel(const __bf16* __restrict__ A, int lda,
                                 const __bf16* __restrict__ Bt, int ldb,
                                 float* __restrict__ C, int ldc, int K,
                                 const float* __restrict__ bias) {
  const int lane  = threadIdx.x & 31;
  const int colw  = lane & 15;
  const int half8 = (lane < 16) ? 0 : 8;
  const int koffA = (lane < 16) ? 0 : 16;  // A lane K base
  const int koffB = (lane < 16) ? 0 : 16;  // B lane K base
  const int i0 = blockIdx.x * 16;
  const int n0 = blockIdx.y * 16;
  const __bf16* arow = A  + (size_t)(i0 + colw) * lda;
  const __bf16* brow = Bt + (size_t)(n0 + colw) * ldb;
  v8f acc = v8f_zero();
  for (int kb = 0; kb < K; kb += 32) {
    U16 a, b;
    a.h[0] = *(const v8bf*)(arow + kb + koffA);
    a.h[1] = *(const v8bf*)(arow + kb + koffA + 8);
    b.h[0] = *(const v8bf*)(brow + kb + koffB);
    b.h[1] = *(const v8bf*)(brow + kb + koffB + 8);
    acc = wmma_bf16(a, b, acc);
  }
  const float bv = bias ? bias[n0 + colw] : 0.f;
  for (int r = 0; r < 8; ++r)
    C[(size_t)(i0 + r + half8) * ldc + n0 + colw] = acc[r] + bv;
}

// ---------------------------------------------------------------------------
// pack: apply rigid frames, build extended Q/K (28->32, scales folded),
// rank-1 logit biases, transposed V and global V-points for B-operand loads.
// ---------------------------------------------------------------------------
__global__ void pack_kernel(const float* __restrict__ qh, const float* __restrict__ kh,
                            const float* __restrict__ vh, const float* __restrict__ qp,
                            const float* __restrict__ kp, const float* __restrict__ vp,
                            const float* __restrict__ rots, const float* __restrict__ trans,
                            const float* __restrict__ gamma,
                            __bf16* __restrict__ Qext, __bf16* __restrict__ Kext,
                            __bf16* __restrict__ Vt,   __bf16* __restrict__ Vgt,
                            float* __restrict__ qbias, float* __restrict__ kbias) {
  int gid = blockIdx.x * blockDim.x + threadIdx.x;
  if (gid >= N_ * H_) return;
  const int i = gid / H_, h = gid % H_;
  const float* R = rots + (size_t)i * 9;
  const float t0 = trans[i * 3 + 0], t1 = trans[i * 3 + 1], t2 = trans[i * 3 + 2];
  const float ch = 0.5f * W_Cc * gamma[h];

  __bf16* Qr = Qext + ((size_t)h * N_ + i) * 32;
  __bf16* Kr = Kext + ((size_t)h * N_ + i) * 32;
  for (int d = 0; d < C_H_; ++d) {
    Qr[d] = (__bf16)(W_Lc * 0.25f * qh[(size_t)i * 192 + h * 16 + d]);
    Kr[d] = (__bf16)(kh[(size_t)i * 192 + h * 16 + d]);
  }
  float qsq = 0.f, ksq = 0.f;
  for (int p = 0; p < PQ_; ++p) {
    const float* yq = qp + (size_t)i * 144 + (h * PQ_ + p) * 3;
    float gx = R[0]*yq[0] + R[1]*yq[1] + R[2]*yq[2] + t0;
    float gy = R[3]*yq[0] + R[4]*yq[1] + R[5]*yq[2] + t1;
    float gz = R[6]*yq[0] + R[7]*yq[1] + R[8]*yq[2] + t2;
    qsq += gx*gx + gy*gy + gz*gz;
    Qr[16 + p*3 + 0] = (__bf16)(W_Lc * 2.f * ch * gx);
    Qr[16 + p*3 + 1] = (__bf16)(W_Lc * 2.f * ch * gy);
    Qr[16 + p*3 + 2] = (__bf16)(W_Lc * 2.f * ch * gz);
    const float* yk = kp + (size_t)i * 144 + (h * PQ_ + p) * 3;
    float hx = R[0]*yk[0] + R[1]*yk[1] + R[2]*yk[2] + t0;
    float hy = R[3]*yk[0] + R[4]*yk[1] + R[5]*yk[2] + t1;
    float hz = R[6]*yk[0] + R[7]*yk[1] + R[8]*yk[2] + t2;
    ksq += hx*hx + hy*hy + hz*hz;
    Kr[16 + p*3 + 0] = (__bf16)hx;
    Kr[16 + p*3 + 1] = (__bf16)hy;
    Kr[16 + p*3 + 2] = (__bf16)hz;
  }
  for (int d = 28; d < 32; ++d) { Qr[d] = (__bf16)0.f; Kr[d] = (__bf16)0.f; }
  qbias[h * N_ + i] = -W_Lc * ch * qsq;
  kbias[h * N_ + i] = -W_Lc * ch * ksq;

  for (int d = 0; d < C_H_; ++d)
    Vt[((size_t)h * 16 + d) * N_ + i] = (__bf16)vh[(size_t)i * 192 + h * 16 + d];
  for (int p = 0; p < PV_; ++p) {
    const float* yv = vp + (size_t)i * 288 + (h * PV_ + p) * 3;
    float gx = R[0]*yv[0] + R[1]*yv[1] + R[2]*yv[2] + t0;
    float gy = R[3]*yv[0] + R[4]*yv[1] + R[5]*yv[2] + t1;
    float gz = R[6]*yv[0] + R[7]*yv[1] + R[8]*yv[2] + t2;
    Vgt[((size_t)h * 32 + p*3 + 0) * N_ + i] = (__bf16)gx;
    Vgt[((size_t)h * 32 + p*3 + 1) * N_ + i] = (__bf16)gy;
    Vgt[((size_t)h * 32 + p*3 + 2) * N_ + i] = (__bf16)gz;
  }
  for (int c = 24; c < 32; ++c) Vgt[((size_t)h * 32 + c) * N_ + i] = (__bf16)0.f;
}

// ---------------------------------------------------------------------------
// b_ij = W_L * z @ Wb  ->  bbuf[h][i][j]   (first z pass; cvt->bf16 in regs)
// one wave per (i, 16-wide j tile); M = 16 j-rows, N = heads(pad16), K = 128.
// ---------------------------------------------------------------------------
__global__ void bij_kernel(const float* __restrict__ z,
                           const __bf16* __restrict__ WbT,
                           float* __restrict__ bbuf) {
  const int lane  = threadIdx.x & 31;
  const int wave  = threadIdx.x >> 5;
  const int gid   = blockIdx.x * 8 + wave;   // 512 * 32 tiles
  const int i     = gid >> 5;
  const int j0    = (gid & 31) * 16;
  const int colw  = lane & 15;
  const int half8 = (lane < 16) ? 0 : 8;
  const int koffA = (lane < 16) ? 0 : 16;
  const int koffB = (lane < 16) ? 0 : 16;
  const float*  zrow = z + ((size_t)i * N_ + j0 + colw) * C_Z_;
  const __bf16* brow = WbT + colw * C_Z_;
  v8f acc = v8f_zero();
  for (int kb = 0; kb < C_Z_; kb += 32) {
    U16 a, b;
    const float* pa = zrow + kb + koffA;
    for (int t = 0; t < 8; ++t) a.v[t]     = (__bf16)pa[t];
    for (int t = 0; t < 8; ++t) a.v[8 + t] = (__bf16)pa[8 + t];
    b.h[0] = *(const v8bf*)(brow + kb + koffB);
    b.h[1] = *(const v8bf*)(brow + kb + koffB + 8);
    acc = wmma_bf16(a, b, acc);
  }
  if (colw < H_) {
    float* dst = bbuf + (size_t)colw * N_ * N_ + (size_t)i * N_;
    for (int r = 0; r < 8; ++r) dst[j0 + r + half8] = acc[r];
  }
}

// ---------------------------------------------------------------------------
// fused flash-style IPA attention.
// block = 12 waves (one head each), 16 query rows per block, j tiles of 32.
// per tile: TDM stages z slab into LDS (overlaps logits), logits via WMMA
// (K=32 ext-QK), online softmax, bf16 probs to LDS, then o/o_hp (M=i per
// head) and o_hat (M=heads per query row, B operands from the LDS z slab).
// ---------------------------------------------------------------------------
__global__ void __launch_bounds__(384)
attn_kernel(const __bf16* __restrict__ Qext, const __bf16* __restrict__ Kext,
            const __bf16* __restrict__ Vt,   const __bf16* __restrict__ Vgt,
            const float* __restrict__ qbias, const float* __restrict__ kbias,
            const float* __restrict__ bbuf,  const float* __restrict__ z,
            __bf16* __restrict__ cat, float* __restrict__ ohp_raw) {
  __shared__ float  zbuf[16 * 32 * 128];  // 256 KB TDM-staged z slab [i][j][d]
  __shared__ __bf16 Pl[16 * 16 * 32];     // 16 KB probs [i_local][h(pad16)][j32]
  __shared__ float  corrS[16 * 16];       // [h][i_local] running-max correction
  __shared__ float  linvS[16 * 16];       // [h][i_local] 1/l

  const int tid   = threadIdx.x;
  const int lane  = tid & 31;
  const int h     = tid >> 5;             // wave == head
  const int colw  = lane & 15;
  const int half8 = (lane < 16) ? 0 : 8;
  const int koffA = (lane < 16) ? 0 : 16;
  const int koffB = (lane < 16) ? 0 : 16;
  const int i0    = blockIdx.x * 16;
  const unsigned zbuf_lds = (unsigned)(uintptr_t)(void*)&zbuf[0];

  for (int idx = tid; idx < 16 * 16 * 32; idx += blockDim.x) Pl[idx] = (__bf16)0.f;
  if (tid < 256) { corrS[tid] = 1.f; linvS[tid] = 1.f; }
  __syncthreads();

  U16 aq;
  {
    const __bf16* qr = Qext + ((size_t)h * N_ + i0 + colw) * 32;
    aq.h[0] = *(const v8bf*)(qr + koffA);
    aq.h[1] = *(const v8bf*)(qr + koffA + 8);
  }
  float qb8[8];
  for (int r = 0; r < 8; ++r) qb8[r] = qbias[h * N_ + i0 + r + half8];

  float m_run[8], l_run[8];
  for (int r = 0; r < 8; ++r) { m_run[r] = -1e30f; l_run[r] = 0.f; }

  v8f o_acc = v8f_zero(), ohp0 = v8f_zero(), ohp1 = v8f_zero();
  const int nown = (h < 4) ? 2 : 1;       // 16 query rows over 12 waves
  v8f ohat[2][8];
  for (int e = 0; e < 2; ++e)
    for (int nb = 0; nb < 8; ++nb) ohat[e][nb] = v8f_zero();

  for (int t = 0; t < 16; ++t) {
    const int j0 = t * 32;
    // ---- kick off TDM for this tile's z slab (overlaps logits) --------
    if (h == 0)
      tdm_load_z_slab(z + ((size_t)i0 * N_ + j0) * C_Z_, zbuf_lds);
    // ---- logits: two 16x16 WMMAs (K=32) -------------------------------
    v8f s0 = v8f_zero(), s1 = v8f_zero();
    {
      U16 bk;
      const __bf16* kr = Kext + ((size_t)h * N_ + j0 + colw) * 32;
      bk.h[0] = *(const v8bf*)(kr + koffB);
      bk.h[1] = *(const v8bf*)(kr + koffB + 8);
      s0 = wmma_bf16(aq, bk, s0);
      kr += 16 * 32;
      bk.h[0] = *(const v8bf*)(kr + koffB);
      bk.h[1] = *(const v8bf*)(kr + koffB + 8);
      s1 = wmma_bf16(aq, bk, s1);
    }
    const float kb0 = kbias[h * N_ + j0 + colw];
    const float kb1 = kbias[h * N_ + j0 + 16 + colw];
    const float* brow = bbuf + (size_t)h * N_ * N_ + (size_t)i0 * N_;
    if (t + 1 < 16)
      __builtin_prefetch(brow + (size_t)half8 * N_ + j0 + 32, 0, 0);
    for (int r = 0; r < 8; ++r) {
      const float* bij = brow + (size_t)(r + half8) * N_ + j0 + colw;
      s0[r] += bij[0]  + qb8[r] + kb0;
      s1[r] += bij[16] + qb8[r] + kb1;
    }
    // ---- online softmax over this 32-wide tile ------------------------
    for (int r = 0; r < 8; ++r) {
      float mx = fmaxf(s0[r], s1[r]);
      for (int d = 1; d < 16; d <<= 1) mx = fmaxf(mx, __shfl_xor(mx, d, 32));
      const float mnew = fmaxf(m_run[r], mx);
      const float cr   = __expf(m_run[r] - mnew);
      const float p0   = __expf(s0[r] - mnew);
      const float p1   = __expf(s1[r] - mnew);
      float ls = p0 + p1;
      for (int d = 1; d < 16; d <<= 1) ls += __shfl_xor(ls, d, 32);
      l_run[r] = l_run[r] * cr + ls;
      m_run[r] = mnew;
      o_acc[r] *= cr; ohp0[r] *= cr; ohp1[r] *= cr;
      if (colw == 0) corrS[h * 16 + r + half8] = cr;
      __bf16* pw = Pl + ((size_t)(r + half8) * 16 + h) * 32;
      pw[colw]      = (__bf16)p0;
      pw[16 + colw] = (__bf16)p1;
    }
    // ---- wait for TDM slab, publish probs -----------------------------
    if (h == 0) __builtin_amdgcn_s_wait_tensorcnt(0);
    __syncthreads();

    // ---- o / o_hp: per head, M = i rows, K = 32 j ---------------------
    {
      U16 ap, bv;
      const __bf16* pr = Pl + ((size_t)colw * 16 + h) * 32;
      ap.h[0] = *(const v8bf*)(pr + koffA);
      ap.h[1] = *(const v8bf*)(pr + koffA + 8);
      const __bf16* vr = Vt + ((size_t)h * 16 + colw) * N_ + j0 + koffB;
      bv.h[0] = *(const v8bf*)(vr);
      bv.h[1] = *(const v8bf*)(vr + 8);
      o_acc = wmma_bf16(ap, bv, o_acc);
      const __bf16* g0 = Vgt + ((size_t)h * 32 + colw) * N_ + j0 + koffB;
      bv.h[0] = *(const v8bf*)(g0);
      bv.h[1] = *(const v8bf*)(g0 + 8);
      ohp0 = wmma_bf16(ap, bv, ohp0);
      const __bf16* g1 = Vgt + ((size_t)h * 32 + 16 + colw) * N_ + j0 + koffB;
      bv.h[0] = *(const v8bf*)(g1);
      bv.h[1] = *(const v8bf*)(g1 + 8);
      ohp1 = wmma_bf16(ap, bv, ohp1);
    }
    // ---- o_hat: per owned query row, M = heads, B from LDS z slab -----
    for (int e = 0; e < nown; ++e) {
      const int il = h + 12 * e;
      float c8[8];
      for (int r = 0; r < 8; ++r) c8[r] = corrS[(r + half8) * 16 + il];
      for (int nb = 0; nb < 8; ++nb)
        for (int r = 0; r < 8; ++r) ohat[e][nb][r] *= c8[r];
      U16 ap;
      const __bf16* pr = Pl + ((size_t)il * 16 + colw) * 32;
      ap.h[0] = *(const v8bf*)(pr + koffA);
      ap.h[1] = *(const v8bf*)(pr + koffA + 8);
      const float* zsl = zbuf + ((size_t)il * 32 + koffB) * C_Z_;
      for (int nb = 0; nb < 8; ++nb) {
        U16 bz;
        const float* zc = zsl + nb * 16 + colw;
        for (int kk = 0; kk < 16; ++kk) bz.v[kk] = (__bf16)zc[(size_t)kk * C_Z_];
        ohat[e][nb] = wmma_bf16(ap, bz, ohat[e][nb]);
      }
    }
    __syncthreads();   // z slab + Pl free for next tile
  }

  // ---- finalize: divide by l, emit cat pieces --------------------------
  float linv[8];
  for (int r = 0; r < 8; ++r) {
    linv[r] = 1.f / l_run[r];
    if (colw == 0) linvS[h * 16 + r + half8] = linv[r];
  }
  for (int r = 0; r < 8; ++r) {
    const int irow = i0 + r + half8;
    cat[(size_t)irow * CAT_W + 1536 + h * 16 + colw] = (__bf16)(o_acc[r] * linv[r]);
    ohp_raw[((size_t)irow * H_ + h) * 24 + colw] = ohp0[r] * linv[r];
    if (colw < 8)
      ohp_raw[((size_t)irow * H_ + h) * 24 + 16 + colw] = ohp1[r] * linv[r];
  }
  __syncthreads();
  for (int e = 0; e < nown; ++e) {
    const int il = h + 12 * e;
    float c8[8];
    for (int r = 0; r < 8; ++r) c8[r] = linvS[(r + half8) * 16 + il];
    for (int nb = 0; nb < 8; ++nb)
      for (int r = 0; r < 8; ++r) {
        const int hrow = r + half8;
        if (hrow < H_)
          cat[(size_t)(i0 + il) * CAT_W + hrow * 128 + nb * 16 + colw] =
              (__bf16)(ohat[e][nb][r] * c8[r]);
      }
  }
}

// ---------------------------------------------------------------------------
// inverse rigid transform on o_hp, write o_hp + norms into cat
// ---------------------------------------------------------------------------
__global__ void ohp_fin_kernel(const float* __restrict__ ohp_raw,
                               const float* __restrict__ rots,
                               const float* __restrict__ trans,
                               __bf16* __restrict__ cat) {
  int gid = blockIdx.x * blockDim.x + threadIdx.x;
  if (gid >= N_ * H_ * PV_) return;
  const int i = gid / (H_ * PV_);
  const int rem = gid % (H_ * PV_);
  const int h = rem / PV_, p = rem % PV_;
  const float* v = ohp_raw + ((size_t)i * H_ + h) * 24 + p * 3;
  const float* R = rots + (size_t)i * 9;
  const float d0 = v[0] - trans[i * 3 + 0];
  const float d1 = v[1] - trans[i * 3 + 1];
  const float d2 = v[2] - trans[i * 3 + 2];
  const float l0 = R[0]*d0 + R[3]*d1 + R[6]*d2;   // R^T (v - t)
  const float l1 = R[1]*d0 + R[4]*d1 + R[7]*d2;
  const float l2 = R[2]*d0 + R[5]*d1 + R[8]*d2;
  __bf16* c = cat + (size_t)i * CAT_W;
  c[1728 + h * 24 + p * 3 + 0] = (__bf16)l0;
  c[1728 + h * 24 + p * 3 + 1] = (__bf16)l1;
  c[1728 + h * 24 + p * 3 + 2] = (__bf16)l2;
  c[2016 + h * 8 + p] = (__bf16)sqrtf(l0*l0 + l1*l1 + l2*l2);
}

// ---------------------------------------------------------------------------
extern "C" void kernel_launch(void* const* d_in, const int* in_sizes, int n_in,
                              void* d_out, int out_size, void* d_ws, size_t ws_size,
                              hipStream_t stream) {
  const float* s_i   = (const float*)d_in[0];
  const float* z_ij  = (const float*)d_in[1];
  const float* rots  = (const float*)d_in[2];
  const float* trans = (const float*)d_in[3];
  const float* Wq    = (const float*)d_in[4];
  const float* Wk    = (const float*)d_in[5];
  const float* Wv    = (const float*)d_in[6];
  const float* Wqp   = (const float*)d_in[7];
  const float* Wkp   = (const float*)d_in[8];
  const float* Wvp   = (const float*)d_in[9];
  const float* Wb    = (const float*)d_in[10];
  const float* gamma = (const float*)d_in[11];
  const float* Wout  = (const float*)d_in[12];
  const float* bout  = (const float*)d_in[13];
  float* out = (float*)d_out;

  char* ws = (char*)d_ws;
  size_t off = 0;
  auto alloc = [&](size_t bytes) -> void* {
    void* p = ws + off;
    off = (off + bytes + 255) & ~(size_t)255;
    return p;
  };

  __bf16* s_bf  = (__bf16*)alloc((size_t)N_ * C_S_ * 2);
  __bf16* WqT   = (__bf16*)alloc((size_t)192 * C_S_ * 2);
  __bf16* WkT   = (__bf16*)alloc((size_t)192 * C_S_ * 2);
  __bf16* WvT   = (__bf16*)alloc((size_t)192 * C_S_ * 2);
  __bf16* WqpT  = (__bf16*)alloc((size_t)144 * C_S_ * 2);
  __bf16* WkpT  = (__bf16*)alloc((size_t)144 * C_S_ * 2);
  __bf16* WvpT  = (__bf16*)alloc((size_t)288 * C_S_ * 2);
  __bf16* WbT   = (__bf16*)alloc((size_t)16 * C_Z_ * 2);
  __bf16* WoutT = (__bf16*)alloc((size_t)384 * CAT_W * 2);
  float*  qh    = (float*)alloc((size_t)N_ * 192 * 4);
  float*  kh    = (float*)alloc((size_t)N_ * 192 * 4);
  float*  vh    = (float*)alloc((size_t)N_ * 192 * 4);
  float*  qp    = (float*)alloc((size_t)N_ * 144 * 4);
  float*  kp    = (float*)alloc((size_t)N_ * 144 * 4);
  float*  vp    = (float*)alloc((size_t)N_ * 288 * 4);
  __bf16* Qext  = (__bf16*)alloc((size_t)H_ * N_ * 32 * 2);
  __bf16* Kext  = (__bf16*)alloc((size_t)H_ * N_ * 32 * 2);
  float*  qbias = (float*)alloc((size_t)H_ * N_ * 4);
  float*  kbias = (float*)alloc((size_t)H_ * N_ * 4);
  __bf16* Vt    = (__bf16*)alloc((size_t)H_ * 16 * N_ * 2);
  __bf16* Vgt   = (__bf16*)alloc((size_t)H_ * 32 * N_ * 2);
  float*  bbuf  = (float*)alloc((size_t)H_ * N_ * N_ * 4);
  __bf16* cat   = (__bf16*)alloc((size_t)N_ * CAT_W * 2);
  float*  ohp_r = (float*)alloc((size_t)N_ * H_ * 24 * 4);
  (void)ws_size; (void)in_sizes; (void)n_in; (void)out_size;

  // 1) precision prep (bf16 activations, transposed bf16 weights)
  {
    int n = N_ * C_S_;
    cvt_bf16_kernel<<<(n + 255) / 256, 256, 0, stream>>>(s_bf, s_i, n);
  }
  {
    int n = C_S_ * 192;
    transpose_cvt_kernel<<<(n + 255) / 256, 256, 0, stream>>>(WqT, Wq, C_S_, 192);
    transpose_cvt_kernel<<<(n + 255) / 256, 256, 0, stream>>>(WkT, Wk, C_S_, 192);
    transpose_cvt_kernel<<<(n + 255) / 256, 256, 0, stream>>>(WvT, Wv, C_S_, 192);
    n = C_S_ * 144;
    transpose_cvt_kernel<<<(n + 255) / 256, 256, 0, stream>>>(WqpT, Wqp, C_S_, 144);
    transpose_cvt_kernel<<<(n + 255) / 256, 256, 0, stream>>>(WkpT, Wkp, C_S_, 144);
    n = C_S_ * 288;
    transpose_cvt_kernel<<<(n + 255) / 256, 256, 0, stream>>>(WvpT, Wvp, C_S_, 288);
    n = CAT_W * 384;
    transpose_cvt_kernel<<<(n + 255) / 256, 256, 0, stream>>>(WoutT, Wout, CAT_W, 384);
    prep_wb_kernel<<<(16 * C_Z_ + 255) / 256, 256, 0, stream>>>(WbT, Wb);
  }
  // 2) projections (WMMA bf16 GEMMs)
  gemm_bf16_kernel<<<dim3(32, 12), 32, 0, stream>>>(s_bf, C_S_, WqT,  C_S_, qh, 192, C_S_, nullptr);
  gemm_bf16_kernel<<<dim3(32, 12), 32, 0, stream>>>(s_bf, C_S_, WkT,  C_S_, kh, 192, C_S_, nullptr);
  gemm_bf16_kernel<<<dim3(32, 12), 32, 0, stream>>>(s_bf, C_S_, WvT,  C_S_, vh, 192, C_S_, nullptr);
  gemm_bf16_kernel<<<dim3(32,  9), 32, 0, stream>>>(s_bf, C_S_, WqpT, C_S_, qp, 144, C_S_, nullptr);
  gemm_bf16_kernel<<<dim3(32,  9), 32, 0, stream>>>(s_bf, C_S_, WkpT, C_S_, kp, 144, C_S_, nullptr);
  gemm_bf16_kernel<<<dim3(32, 18), 32, 0, stream>>>(s_bf, C_S_, WvpT, C_S_, vp, 288, C_S_, nullptr);
  // 3) rigid frames + operand packing
  pack_kernel<<<(N_ * H_ + 255) / 256, 256, 0, stream>>>(
      qh, kh, vh, qp, kp, vp, rots, trans, gamma, Qext, Kext, Vt, Vgt, qbias, kbias);
  // 4) pair bias (WMMA over z, first z pass)
  bij_kernel<<<2048, 256, 0, stream>>>(z_ij, WbT, bbuf);
  // 5) fused flash attention (second z pass, TDM-staged, mostly L2-resident)
  attn_kernel<<<32, 384, 0, stream>>>(Qext, Kext, Vt, Vgt, qbias, kbias,
                                      bbuf, z_ij, cat, ohp_r);
  // 6) inverse frames + norms
  ohp_fin_kernel<<<(N_ * H_ * PV_ + 255) / 256, 256, 0, stream>>>(ohp_r, rots, trans, cat);
  // 7) output projection (K = 2112 = 66 * 32) + bias
  gemm_bf16_kernel<<<dim3(32, 24), 32, 0, stream>>>(cat, CAT_W, WoutT, CAT_W,
                                                    out, 384, CAT_W, bout);
}
